// MultiHeadedAttention_14826227106535
// MI455X (gfx1250) — matrix operand order repaired and checked by reading, not compiled, not run on hardware
//
#include <hip/hip_runtime.h>

// MultiHeadedAttention for MI455X (gfx1250, wave32, WMMA bf16 path).
// Pipeline: f32->bf16 convert, weight transposes, QKV GEMM (WMMA, software-
// pipelined 32x64 wave tiles) with fused bias + 1/sqrt(Hd) Q-scale + V
// transpose, causal flash attention (WMMA scores + online softmax + WMMA PV,
// batched fragment loads + global prefetch), output GEMM (WMMA) + bias.

#define S_LEN 4096
#define D_MODEL 1024
#define N_HEADS 16
#define H_DIM 64

typedef __bf16 v16bf __attribute__((ext_vector_type(16)));
typedef __bf16 bf16x8 __attribute__((ext_vector_type(8)));
typedef float  v8f   __attribute__((ext_vector_type(8)));

// ---- WMMA fragment helpers (layouts per CDNA5 ISA 7.12.2) ----------------
// A 16x32 bf16: lane = M + 16*half; half 0 holds K {k0..k0+7, k0+16..k0+23},
// half 1 holds K {k0+8..k0+15, k0+24..k0+31}. Elements ascend with VGPR idx.
static __device__ inline v16bf load_a_frag(const __bf16* base, int ld, int k0, int lane) {
  int m    = lane & 15;
  int koff = (lane >> 4) * 8;
  const __bf16* p = base + m * ld + k0 + koff;
  bf16x8 lo = *reinterpret_cast<const bf16x8*>(p);
  bf16x8 hi = *reinterpret_cast<const bf16x8*>(p + 16);
  v16bf r;
#pragma unroll
  for (int i = 0; i < 8; ++i) { r[i] = lo[i]; r[i + 8] = hi[i]; }
  return r;
}

// B 32x16 bf16 from B-transposed [N,K] row-major: lane = N + 16*half,
// half 0 holds K {k0..k0+7, k0+16..k0+23}, half 1 holds K {k0+8..k0+15, k0+24..k0+31}.
static __device__ inline v16bf load_b_frag(const __bf16* base, int ld, int k0, int lane) {
  int n  = lane & 15;
  int kb = (lane >> 4) * 8;
  const __bf16* p = base + n * ld + k0 + kb;
  bf16x8 lo = *reinterpret_cast<const bf16x8*>(p);
  bf16x8 hi = *reinterpret_cast<const bf16x8*>(p + 16);
  v16bf r;
#pragma unroll
  for (int i = 0; i < 8; ++i) { r[i] = lo[i]; r[i + 8] = hi[i]; }
  return r;
}

static __device__ inline v8f wmma_bf16(v16bf a, v16bf b, v8f c) {
  return __builtin_amdgcn_wmma_f32_16x16x32_bf16(false, a, false, b,
                                                 (short)0, c, false, false);
}

// C 16x16 f32 layout: element (m,n): lane = n + 16*(m/8), vgpr j = m%8.

// One wave computes a 32(M) x 64(N) tile, K-loop step 32, register
// double-buffered so next-tile loads overlap current-tile WMMAs.
// c[f] = row-tile 0, c[4+f] = row-tile 1.
static __device__ inline void gemm_wave_32x64(const __bf16* A, int lda,
                                              const __bf16* Bt, int ldb,
                                              int K, int lane, v8f c[8]) {
  v16bf a0 = load_a_frag(A, lda, 0, lane);
  v16bf a1 = load_a_frag(A + 16 * lda, lda, 0, lane);
  v16bf b[4];
#pragma unroll
  for (int f = 0; f < 4; ++f) b[f] = load_b_frag(Bt + f * 16 * ldb, ldb, 0, lane);

  for (int k0 = 32; k0 < K; k0 += 32) {
    // prefetch next k-tile into registers (issues loads before the waits
    // for the WMMAs below, so VMEM overlaps the matrix pipe)
    v16bf an0 = load_a_frag(A, lda, k0, lane);
    v16bf an1 = load_a_frag(A + 16 * lda, lda, k0, lane);
    v16bf bn[4];
#pragma unroll
    for (int f = 0; f < 4; ++f) bn[f] = load_b_frag(Bt + f * 16 * ldb, ldb, k0, lane);

#pragma unroll
    for (int f = 0; f < 4; ++f) c[f] = wmma_bf16(a0, b[f], c[f]);
#pragma unroll
    for (int f = 0; f < 4; ++f) c[4 + f] = wmma_bf16(a1, b[f], c[4 + f]);

    a0 = an0; a1 = an1;
#pragma unroll
    for (int f = 0; f < 4; ++f) b[f] = bn[f];
  }
  // epilogue tile
#pragma unroll
  for (int f = 0; f < 4; ++f) c[f] = wmma_bf16(a0, b[f], c[f]);
#pragma unroll
  for (int f = 0; f < 4; ++f) c[4 + f] = wmma_bf16(a1, b[f], c[4 + f]);
}

// ---- elementwise convert f32 -> bf16 -------------------------------------
__global__ __launch_bounds__(256) void k_f32_to_bf16(const float* __restrict__ in,
                                                     __bf16* __restrict__ out, int n) {
  int i = (blockIdx.x * blockDim.x + threadIdx.x) * 4;
  if (i < n) {
    float4 v = *reinterpret_cast<const float4*>(in + i);
    out[i + 0] = (__bf16)v.x;
    out[i + 1] = (__bf16)v.y;
    out[i + 2] = (__bf16)v.z;
    out[i + 3] = (__bf16)v.w;
  }
}

// ---- LDS-tiled transpose + convert: out[c][r] = (bf16) in[r][c] ----------
__global__ __launch_bounds__(1024) void k_transpose_f32_bf16(const float* __restrict__ in,
                                                             __bf16* __restrict__ out,
                                                             int rows, int cols) {
  __shared__ float tile[32][33];
  int c = blockIdx.x * 32 + threadIdx.x;
  int r = blockIdx.y * 32 + threadIdx.y;
  tile[threadIdx.y][threadIdx.x] = in[r * cols + c];
  __syncthreads();
  int oc = blockIdx.y * 32 + threadIdx.x;    // output col = input row
  int orow = blockIdx.x * 32 + threadIdx.y;  // output row = input col
  out[orow * rows + oc] = (__bf16)tile[threadIdx.x][threadIdx.y];
}

// ---- QKV projection: QKV = X * Wqkv + b, split into Q (scaled), K, Vt ----
__global__ __launch_bounds__(256) void k_gemm_qkv(const __bf16* __restrict__ A,
                                                  const __bf16* __restrict__ Bt,
                                                  const float* __restrict__ bias,
                                                  __bf16* __restrict__ Qb,
                                                  __bf16* __restrict__ Kb,
                                                  __bf16* __restrict__ Vt) {
  int lane = threadIdx.x & 31;
  int wave = threadIdx.x >> 5;
  int mbase = blockIdx.x * 256 + wave * 32;
  int nbase = blockIdx.y * 64;
  v8f c[8];
#pragma unroll
  for (int f = 0; f < 8; ++f)
#pragma unroll
    for (int j = 0; j < 8; ++j) c[f][j] = 0.0f;

  gemm_wave_32x64(A + mbase * D_MODEL, D_MODEL, Bt + nbase * D_MODEL, D_MODEL,
                  D_MODEL, lane, c);

  int lh = lane >> 4, ln = lane & 15;
#pragma unroll
  for (int r = 0; r < 2; ++r)
#pragma unroll
    for (int f = 0; f < 4; ++f) {
      int n = nbase + f * 16 + ln;
      float bv = bias[n];
#pragma unroll
      for (int j = 0; j < 8; ++j) {
        int m = mbase + r * 16 + 8 * lh + j;
        float v = c[r * 4 + f][j] + bv;
        if (n < D_MODEL) {
          Qb[m * D_MODEL + n] = (__bf16)(v * 0.125f);  // fold 1/sqrt(64)
        } else if (n < 2 * D_MODEL) {
          Kb[m * D_MODEL + (n - D_MODEL)] = (__bf16)v;
        } else {
          Vt[(n - 2 * D_MODEL) * S_LEN + m] = (__bf16)v;  // transposed V
        }
      }
    }
}

// ---- causal flash attention: block = 1 head x 64 rows, 4 waves ------------
__global__ __launch_bounds__(128) void k_flash(const __bf16* __restrict__ Qb,
                                               const __bf16* __restrict__ Kb,
                                               const __bf16* __restrict__ Vt,
                                               __bf16* __restrict__ Ob) {
  __shared__ __bf16 Pl[4][16][64];  // per-wave P tile (rows x 64 keys)
  int lane = threadIdx.x & 31;
  int wave = threadIdx.x >> 5;
  int h = blockIdx.y;
  int rb = blockIdx.x * 64;
  int rwave = rb + wave * 16;
  int lh = lane >> 4, ln = lane & 15;

  // Q fragments for this wave's 16 rows (Hd=64 -> 2 K-tiles), resident all loop.
  const __bf16* qptr = Qb + rwave * D_MODEL + h * H_DIM;
  v16bf aq0 = load_a_frag(qptr, D_MODEL, 0, lane);
  v16bf aq1 = load_a_frag(qptr, D_MODEL, 32, lane);

  v8f acc[4];
  float mrow[8], lrow[8];
#pragma unroll
  for (int f = 0; f < 4; ++f)
#pragma unroll
    for (int j = 0; j < 8; ++j) acc[f][j] = 0.0f;
#pragma unroll
  for (int j = 0; j < 8; ++j) { mrow[j] = -1e30f; lrow[j] = 0.0f; }

  int tend = rb + 64;  // causal: keys up to the block's last row
  for (int t0 = 0; t0 < tend; t0 += 64) {
    // prefetch next key tile (K rows and Vt columns) into cache
    if (t0 + 64 < tend) {
      int pid = threadIdx.x;
      if (pid < 64)
        __builtin_prefetch(Kb + (size_t)(t0 + 64 + pid) * D_MODEL + h * H_DIM, 0, 1);
      else
        __builtin_prefetch(Vt + (size_t)(h * H_DIM + (pid - 64)) * S_LEN + (t0 + 64), 0, 1);
    }

    // batch-load all 8 K-fragments, then run the WMMA chain
    v16bf bk[4][2];
#pragma unroll
    for (int f = 0; f < 4; ++f) {
      const __bf16* kptr = Kb + (t0 + f * 16) * D_MODEL + h * H_DIM;
      bk[f][0] = load_b_frag(kptr, D_MODEL, 0, lane);
      bk[f][1] = load_b_frag(kptr, D_MODEL, 32, lane);
    }
    v8f s[4];
#pragma unroll
    for (int f = 0; f < 4; ++f) {
      v8f z;
#pragma unroll
      for (int j = 0; j < 8; ++j) z[j] = 0.0f;
      z = wmma_bf16(aq0, bk[f][0], z);
      z = wmma_bf16(aq1, bk[f][1], z);
      s[f] = z;
    }

    // causal mask + per-row tile max
    float tmax[8];
#pragma unroll
    for (int j = 0; j < 8; ++j) tmax[j] = -1e30f;
#pragma unroll
    for (int f = 0; f < 4; ++f)
#pragma unroll
      for (int j = 0; j < 8; ++j) {
        int t = t0 + f * 16 + ln;
        int m = rwave + 8 * lh + j;
        float v = (t <= m) ? s[f][j] : -1e30f;
        s[f][j] = v;
        tmax[j] = fmaxf(tmax[j], v);
      }
#pragma unroll
    for (int j = 0; j < 8; ++j)
#pragma unroll
      for (int o = 1; o < 16; o <<= 1)
        tmax[j] = fmaxf(tmax[j], __shfl_xor(tmax[j], o, 32));

    // online softmax update
    float scale[8], rsum[8];
#pragma unroll
    for (int j = 0; j < 8; ++j) {
      float newm = fmaxf(mrow[j], tmax[j]);
      scale[j] = __expf(mrow[j] - newm);
      mrow[j] = newm;
      rsum[j] = 0.0f;
    }
#pragma unroll
    for (int f = 0; f < 4; ++f)
#pragma unroll
      for (int j = 0; j < 8; ++j) {
        float p = __expf(s[f][j] - mrow[j]);
        s[f][j] = p;
        rsum[j] += p;
      }
#pragma unroll
    for (int j = 0; j < 8; ++j)
#pragma unroll
      for (int o = 1; o < 16; o <<= 1)
        rsum[j] += __shfl_xor(rsum[j], o, 32);
#pragma unroll
    for (int j = 0; j < 8; ++j) lrow[j] = lrow[j] * scale[j] + rsum[j];
#pragma unroll
    for (int f = 0; f < 4; ++f)
#pragma unroll
      for (int j = 0; j < 8; ++j) acc[f][j] *= scale[j];

    // P: C-layout -> A-layout via LDS (each wave owns its buffer)
#pragma unroll
    for (int f = 0; f < 4; ++f)
#pragma unroll
      for (int j = 0; j < 8; ++j)
        Pl[wave][8 * lh + j][f * 16 + ln] = (__bf16)s[f][j];
    __syncthreads();

    // batch-load P A-fragments and all 8 V B-fragments, then WMMA chain
    v16bf ap0 = load_a_frag(&Pl[wave][0][0], 64, 0, lane);
    v16bf ap1 = load_a_frag(&Pl[wave][0][0], 64, 32, lane);
    v16bf bv[4][2];
#pragma unroll
    for (int f = 0; f < 4; ++f) {
      const __bf16* vptr = Vt + (h * H_DIM + f * 16) * S_LEN + t0;
      bv[f][0] = load_b_frag(vptr, S_LEN, 0, lane);
      bv[f][1] = load_b_frag(vptr, S_LEN, 32, lane);
    }
#pragma unroll
    for (int f = 0; f < 4; ++f) {
      acc[f] = wmma_bf16(ap0, bv[f][0], acc[f]);
      acc[f] = wmma_bf16(ap1, bv[f][1], acc[f]);
    }
    __syncthreads();
  }

  // normalize and emit bf16 attention output, row-major [S, D]
#pragma unroll
  for (int f = 0; f < 4; ++f)
#pragma unroll
    for (int j = 0; j < 8; ++j) {
      int m = rwave + 8 * lh + j;
      float o = acc[f][j] / lrow[j];
      Ob[m * D_MODEL + h * H_DIM + f * 16 + ln] = (__bf16)o;
    }
}

// ---- output projection: out = attn * Wo + bo (fp32 out) -------------------
__global__ __launch_bounds__(256) void k_gemm_out(const __bf16* __restrict__ A,
                                                  const __bf16* __restrict__ Bt,
                                                  const float* __restrict__ bias,
                                                  float* __restrict__ out) {
  int lane = threadIdx.x & 31;
  int wave = threadIdx.x >> 5;
  int mbase = blockIdx.x * 256 + wave * 32;
  int nbase = blockIdx.y * 64;
  v8f c[8];
#pragma unroll
  for (int f = 0; f < 8; ++f)
#pragma unroll
    for (int j = 0; j < 8; ++j) c[f][j] = 0.0f;

  gemm_wave_32x64(A + mbase * D_MODEL, D_MODEL, Bt + nbase * D_MODEL, D_MODEL,
                  D_MODEL, lane, c);

  int lh = lane >> 4, ln = lane & 15;
#pragma unroll
  for (int r = 0; r < 2; ++r)
#pragma unroll
    for (int f = 0; f < 4; ++f) {
      int n = nbase + f * 16 + ln;
      float bv = bias[n];
#pragma unroll
      for (int j = 0; j < 8; ++j) {
        int m = mbase + r * 16 + 8 * lh + j;
        out[m * D_MODEL + n] = c[r * 4 + f][j] + bv;
      }
    }
}

extern "C" void kernel_launch(void* const* d_in, const int* in_sizes, int n_in,
                              void* d_out, int out_size, void* d_ws, size_t ws_size,
                              hipStream_t stream) {
  (void)in_sizes; (void)n_in; (void)out_size; (void)ws_size;
  const float* input = (const float*)d_in[0];
  const float* Wqkv  = (const float*)d_in[1];
  const float* bqkv  = (const float*)d_in[2];
  const float* Wo    = (const float*)d_in[3];
  const float* bo    = (const float*)d_in[4];
  float* out = (float*)d_out;

  char* ws = (char*)d_ws;
  size_t off = 0;
  auto alloc = [&](size_t bytes) -> char* {
    char* p = ws + off;
    off += (bytes + 255) & ~(size_t)255;
    return p;
  };
  __bf16* Abf   = (__bf16*)alloc((size_t)S_LEN * D_MODEL * 2);       // input bf16
  __bf16* WqkvT = (__bf16*)alloc((size_t)3 * D_MODEL * D_MODEL * 2); // Wqkv^T bf16
  __bf16* WoT   = (__bf16*)alloc((size_t)D_MODEL * D_MODEL * 2);     // Wo^T bf16
  __bf16* Qb    = (__bf16*)alloc((size_t)S_LEN * D_MODEL * 2);       // Q (scaled)
  __bf16* Kb    = (__bf16*)alloc((size_t)S_LEN * D_MODEL * 2);       // K
  __bf16* Vt    = (__bf16*)alloc((size_t)D_MODEL * S_LEN * 2);       // V transposed
  __bf16* Ob    = (__bf16*)alloc((size_t)S_LEN * D_MODEL * 2);       // attn out

  k_f32_to_bf16<<<(S_LEN * D_MODEL / 4 + 255) / 256, 256, 0, stream>>>(
      input, Abf, S_LEN * D_MODEL);
  k_transpose_f32_bf16<<<dim3(3 * D_MODEL / 32, D_MODEL / 32), dim3(32, 32), 0, stream>>>(
      Wqkv, WqkvT, D_MODEL, 3 * D_MODEL);
  k_transpose_f32_bf16<<<dim3(D_MODEL / 32, D_MODEL / 32), dim3(32, 32), 0, stream>>>(
      Wo, WoT, D_MODEL, D_MODEL);
  k_gemm_qkv<<<dim3(S_LEN / 256, 3 * D_MODEL / 64), 256, 0, stream>>>(
      Abf, WqkvT, bqkv, Qb, Kb, Vt);
  k_flash<<<dim3(S_LEN / 64, N_HEADS), 128, 0, stream>>>(Qb, Kb, Vt, Ob);
  k_gemm_out<<<dim3(S_LEN / 256, D_MODEL / 64), 256, 0, stream>>>(Ob, WoT, bo, out);
}
